// LSTMDecoder_31662498906680
// MI455X (gfx1250) — matrix operand order repaired and checked by reading, not compiled, not run on hardware
//
#include <hip/hip_runtime.h>
#include <hip/hip_bf16.h>

#define HID   1024
#define VOC   32000
#define NBATCH 32
#define TSTEPS 128

typedef __attribute__((ext_vector_type(2)))  float  v2f;
typedef __attribute__((ext_vector_type(8)))  float  v8f;
typedef __attribute__((ext_vector_type(8)))  __bf16 v8bf;
typedef __attribute__((ext_vector_type(16))) __bf16 v16bf;

// ---------------- prep kernels ----------------
__global__ void k_add(const float* __restrict__ a, const float* __restrict__ b,
                      float* __restrict__ c, int n) {
    int i = blockIdx.x * blockDim.x + threadIdx.x;
    if (i < n) c[i] = a[i] + b[i];
}

__global__ void k_to_bf16(const float* __restrict__ a, __bf16* __restrict__ b, int n) {
    int i = blockIdx.x * blockDim.x + threadIdx.x;
    if (i < n) b[i] = (__bf16)a[i];
}

__global__ void k_zero(float* __restrict__ p, int n) {
    int i = blockIdx.x * blockDim.x + threadIdx.x;
    if (i < n) p[i] = 0.0f;
}

__device__ __forceinline__ float sigmoidf_(float x) {
    return 1.0f / (1.0f + __expf(-x));
}

// ---------------- LSTM step (fp32 WMMA, split-K=8, fused cell) ----------------
// gates = x @ Wx.T  (+ x2 @ W2.T on the t==0 dual path)  + bias, then pointwise.
// grid: 128 blocks (mtile[0,2) x htile[0,64)), 256 threads = 8 waves.
// Wave w accumulates K-slice [w*128, w*128+128) for all 4 gate tiles of the same
// (m-tile, hidden-column-tile); LDS tree-reduce; 256 threads do the cell update.
__global__ __launch_bounds__(256) void lstm_step(
    const float* __restrict__ x,      // [32,H]  A-matrix 1
    const float* __restrict__ Wx,     // [4H,H]  row-major
    const float* __restrict__ x2,     // [32,H]  A-matrix 2 (t==0 only) or null
    const float* __restrict__ W2,     // [4H,H]
    const float* __restrict__ bias,   // [4H]
    float*       __restrict__ mcell,  // [32,H]  cell state (in/out, in-place)
    float*       __restrict__ hout,   // [32,H]  new hidden
    __bf16*      __restrict__ seq,    // [T*32,H] bf16 copy for projection
    int tstep)
{
    __shared__ float red[8 * 4 * 256];   // [wave][gate][r*32+lane] = 32 KB

    const int tid   = threadIdx.x;
    const int wave  = tid >> 5;
    const int lane  = tid & 31;
    const int mtile = blockIdx.x >> 6;   // 0..1
    const int htile = blockIdx.x & 63;   // 0..63
    const int m0    = mtile * 16;
    const int lm    = lane & 15;
    const int kgrp  = lane >> 4;         // lanes 16-31 hold K+2,K+3 (ISA 16x4 A layout)
    const int ncol  = htile * 16 + lm;   // hidden column handled by this lane (B side)

    const int kb = wave * (HID / 8) + (kgrp << 1);  // this lane's base K

    // Base pointers computed ONCE; loop uses small constant offsets so the
    // compiler emits immediate-offset global_load clauses (no per-iter U64 math).
    const float* pA = x  + (size_t)(m0 + lm) * HID + kb;
    const float* p0 = Wx + (size_t)(0 * HID + ncol) * HID + kb;
    const float* p1 = Wx + (size_t)(1 * HID + ncol) * HID + kb;
    const float* p2 = Wx + (size_t)(2 * HID + ncol) * HID + kb;
    const float* p3 = Wx + (size_t)(3 * HID + ncol) * HID + kb;

    v8f acc0 = {}, acc1 = {}, acc2 = {}, acc3 = {};

    if (x2) {  // t == 0: dual matmul (branch hoisted out of the hot loop)
        const float* qA = x2 + (size_t)(m0 + lm) * HID + kb;
        const float* q0 = W2 + (size_t)(0 * HID + ncol) * HID + kb;
        const float* q1 = W2 + (size_t)(1 * HID + ncol) * HID + kb;
        const float* q2 = W2 + (size_t)(2 * HID + ncol) * HID + kb;
        const float* q3 = W2 + (size_t)(3 * HID + ncol) * HID + kb;
        #pragma unroll 4
        for (int i = 0; i < (HID / 8) / 4; ++i) {
            const int o = i * 4;
            v2f a  = *(const v2f*)(pA + o);
            v2f b0 = *(const v2f*)(p0 + o);
            v2f b1 = *(const v2f*)(p1 + o);
            v2f b2 = *(const v2f*)(p2 + o);
            v2f b3 = *(const v2f*)(p3 + o);
            v2f a2 = *(const v2f*)(qA + o);
            v2f c0 = *(const v2f*)(q0 + o);
            v2f c1 = *(const v2f*)(q1 + o);
            v2f c2 = *(const v2f*)(q2 + o);
            v2f c3 = *(const v2f*)(q3 + o);
            acc0 = __builtin_amdgcn_wmma_f32_16x16x4_f32(false, a,  false, b0, (short)0, acc0, false, false);
            acc1 = __builtin_amdgcn_wmma_f32_16x16x4_f32(false, a,  false, b1, (short)0, acc1, false, false);
            acc2 = __builtin_amdgcn_wmma_f32_16x16x4_f32(false, a,  false, b2, (short)0, acc2, false, false);
            acc3 = __builtin_amdgcn_wmma_f32_16x16x4_f32(false, a,  false, b3, (short)0, acc3, false, false);
            acc0 = __builtin_amdgcn_wmma_f32_16x16x4_f32(false, a2, false, c0, (short)0, acc0, false, false);
            acc1 = __builtin_amdgcn_wmma_f32_16x16x4_f32(false, a2, false, c1, (short)0, acc1, false, false);
            acc2 = __builtin_amdgcn_wmma_f32_16x16x4_f32(false, a2, false, c2, (short)0, acc2, false, false);
            acc3 = __builtin_amdgcn_wmma_f32_16x16x4_f32(false, a2, false, c3, (short)0, acc3, false, false);
        }
    } else {   // t >= 1: single matmul against W_comb
        #pragma unroll 4
        for (int i = 0; i < (HID / 8) / 4; ++i) {
            const int o = i * 4;
            v2f a  = *(const v2f*)(pA + o);
            v2f b0 = *(const v2f*)(p0 + o);
            v2f b1 = *(const v2f*)(p1 + o);
            v2f b2 = *(const v2f*)(p2 + o);
            v2f b3 = *(const v2f*)(p3 + o);
            acc0 = __builtin_amdgcn_wmma_f32_16x16x4_f32(false, a, false, b0, (short)0, acc0, false, false);
            acc1 = __builtin_amdgcn_wmma_f32_16x16x4_f32(false, a, false, b1, (short)0, acc1, false, false);
            acc2 = __builtin_amdgcn_wmma_f32_16x16x4_f32(false, a, false, b2, (short)0, acc2, false, false);
            acc3 = __builtin_amdgcn_wmma_f32_16x16x4_f32(false, a, false, b3, (short)0, acc3, false, false);
        }
    }

    #pragma unroll
    for (int r = 0; r < 8; ++r) {
        red[(wave * 4 + 0) * 256 + r * 32 + lane] = acc0[r];
        red[(wave * 4 + 1) * 256 + r * 32 + lane] = acc1[r];
        red[(wave * 4 + 2) * 256 + r * 32 + lane] = acc2[r];
        red[(wave * 4 + 3) * 256 + r * 32 + lane] = acc3[r];
    }
    __syncthreads();

    // Thread t owns element (r = t>>5, l2 = t&31) of the 16x16 tile for ALL 4 gates:
    // m = r + 8*(l2>=16), n_local = l2&15  (ISA C/D f32 layout)
    const int rr = tid >> 5;
    const int l2 = tid & 31;
    float s[4];
    #pragma unroll
    for (int g = 0; g < 4; ++g) {
        float v = 0.0f;
        #pragma unroll
        for (int w = 0; w < 8; ++w) v += red[(w * 4 + g) * 256 + rr * 32 + l2];
        s[g] = v;
    }
    const int bm = m0 + rr + ((l2 >> 4) << 3);      // batch row 0..31
    const int hc = htile * 16 + (l2 & 15);          // hidden column 0..1023

    const float gi = sigmoidf_(s[0] + bias[0 * HID + hc]);
    const float gf = sigmoidf_(s[1] + bias[1 * HID + hc]);
    const float gg = tanhf   (s[2] + bias[2 * HID + hc]);
    const float go = sigmoidf_(s[3] + bias[3 * HID + hc]);

    const size_t idx = (size_t)bm * HID + hc;
    const float mnew = gf * mcell[idx] + gi * gg;
    const float hnew = go * tanhf(mnew);
    mcell[idx] = mnew;
    hout[idx]  = hnew;
    seq[((size_t)tstep * NBATCH + bm) * HID + hc] = (__bf16)hnew;
}

// ---------------- projection: [4096,1024] x [1024,32000] bf16 WMMA ----------------
// Register-blocked 4 M-tiles x 4 N-tiles per wave (16 accumulators, 16 WMMA per
// K=32 step). Cuts L2 traffic 4x vs 1 M-tile/wave: B = 64x62.5MB, A = 500x8MB.
// grid: 64*500 = 32000 blocks x 128 threads = 128000 waves (mg 0..255).
__global__ __launch_bounds__(128) void proj(
    const __bf16* __restrict__ A,     // [T*32, H] bf16 (row m = t*32 + b)
    const __bf16* __restrict__ B,     // [VOC, H] bf16 (W_lin)
    const float*  __restrict__ blin,  // [VOC]
    float*        __restrict__ out)   // [B, T, VOC]
{
    const int tid  = threadIdx.x;
    const int lane = tid & 31;
    const int lm   = lane & 15;
    const int grp  = lane >> 4;
    const int NG   = (VOC / 16) / 4;              // 500 N-groups
    const int wid  = blockIdx.x * 4 + (tid >> 5); // 0..127999
    const int mg   = wid / NG;                    // 0..255  (rows mg*64 .. +63)
    const int ng   = wid % NG;                    // 0..499 (cols ng*64 .. +63)

    const __bf16* a0 = A + (size_t)(mg * 64 + lm) * HID + grp * 16;
    const __bf16* b0 = B + (size_t)(ng * 64 + lm) * HID + grp * 16;

    v8f acc[4][4] = {};

    for (int k = 0; k < HID; k += 32) {
        if (k + 128 < HID) {
            __builtin_prefetch(a0 + k + 128, 0, 0);
            __builtin_prefetch(b0 + k + 128, 0, 0);
        }
        // A fragments: lanes 0-15 hold K k..k+15; lanes 16-31 K k+16..k+31
        v16bf af[4];
        #pragma unroll
        for (int mt = 0; mt < 4; ++mt) {
            const __bf16* ap = a0 + (size_t)mt * 16 * HID + k;
            v8bf alo = *(const v8bf*)(ap);
            v8bf ahi = *(const v8bf*)(ap + 8);
            af[mt] = __builtin_shufflevector(alo, ahi,
                0, 1, 2, 3, 4, 5, 6, 7, 8, 9, 10, 11, 12, 13, 14, 15);
        }
        #pragma unroll
        for (int nt = 0; nt < 4; ++nt) {
            // B fragment: lanes 0-15 hold K k..k+15 of column n; lanes 16-31 K k+16..k+31
            v16bf bf = *(const v16bf*)(b0 + (size_t)nt * 16 * HID + k);
            #pragma unroll
            for (int mt = 0; mt < 4; ++mt) {
                acc[mt][nt] = __builtin_amdgcn_wmma_f32_16x16x32_bf16(
                    false, af[mt], false, bf, (short)0, acc[mt][nt], false, false);
            }
        }
    }

    #pragma unroll
    for (int nt = 0; nt < 4; ++nt) {
        const int n  = ng * 64 + nt * 16 + lm;
        const float bl = blin[n];
        #pragma unroll
        for (int mt = 0; mt < 4; ++mt) {
            #pragma unroll
            for (int r = 0; r < 8; ++r) {
                const int m = mg * 64 + mt * 16 + r + (grp << 3); // seq row = t*32 + b
                const int t = m >> 5;
                const int b = m & 31;
                out[((size_t)b * TSTEPS + t) * VOC + n] = acc[mt][nt][r] + bl;
            }
        }
    }
}

// ---------------- launch ----------------
extern "C" void kernel_launch(void* const* d_in, const int* in_sizes, int n_in,
                              void* d_out, int out_size, void* d_ws, size_t ws_size,
                              hipStream_t stream) {
    const float* h_in = (const float*)d_in[0];
    const float* x0   = (const float*)d_in[1];
    const float* Wih  = (const float*)d_in[2];
    const float* Whh  = (const float*)d_in[3];
    const float* bih  = (const float*)d_in[4];
    const float* bhh  = (const float*)d_in[5];
    const float* Wlin = (const float*)d_in[6];
    const float* blin = (const float*)d_in[7];
    float* out = (float*)d_out;

    char* p = (char*)d_ws;
    float*  Wc    = (float*)p;  p += (size_t)4 * HID * HID * 4;         // 16 MB
    float*  bias  = (float*)p;  p += (size_t)4 * HID * 4;               // 16 KB
    float*  mcell = (float*)p;  p += (size_t)NBATCH * HID * 4;          // 128 KB
    float*  hA    = (float*)p;  p += (size_t)NBATCH * HID * 4;          // 128 KB
    float*  hB    = (float*)p;  p += (size_t)NBATCH * HID * 4;          // 128 KB
    __bf16* seq   = (__bf16*)p; p += (size_t)TSTEPS * NBATCH * HID * 2; // 8 MB
    __bf16* Wlb   = (__bf16*)p; p += (size_t)VOC * HID * 2;             // 62.5 MB
    (void)ws_size; (void)in_sizes; (void)n_in; (void)out_size;

    const int nW = 4 * HID * HID;
    k_add<<<(nW + 255) / 256, 256, 0, stream>>>(Wih, Whh, Wc, nW);
    k_add<<<(4 * HID + 255) / 256, 256, 0, stream>>>(bih, bhh, bias, 4 * HID);
    k_zero<<<(NBATCH * HID + 255) / 256, 256, 0, stream>>>(mcell, NBATCH * HID);
    const int nL = VOC * HID;
    k_to_bf16<<<(nL + 255) / 256, 256, 0, stream>>>(Wlin, Wlb, nL);

    // t = 0: gates = x0 @ W_ih.T + h @ W_hh.T
    lstm_step<<<128, 256, 0, stream>>>(x0, Wih, h_in, Whh, bias, mcell, hA, seq, 0);
    // t >= 1: x == h, so gates = h @ (W_ih + W_hh).T
    for (int t = 1; t < TSTEPS; ++t) {
        const float* hi = (t & 1) ? hA : hB;
        float*       ho = (t & 1) ? hB : hA;
        lstm_step<<<128, 256, 0, stream>>>(hi, Wc, nullptr, nullptr, bias, mcell, ho, seq, t);
    }

    proj<<<(64 * 500), 128, 0, stream>>>(seq, Wlb, blin, out);
}
